// BitNetLlamaAttention_51230369907331
// MI455X (gfx1250) — 16-bit and 8-bit matrix operand order repaired and checked by reading, compiled, not run on hardware
//
#include <hip/hip_runtime.h>

typedef __attribute__((ext_vector_type(8)))  int          v8i;
typedef __attribute__((ext_vector_type(8)))  float        v8f;
typedef __attribute__((ext_vector_type(8)))  unsigned int v8u;
typedef __attribute__((ext_vector_type(16))) __bf16       v16bf;

#define HID   4096
#define TOK   4096
#define SEQ   2048
#define NHEAD 32
#define NKV   8
#define DHEAD 128
#define KVW   1024

__global__ __launch_bounds__(256)
void abs_partial_kernel(const float* __restrict__ W, long n, float* __restrict__ partial) {
    __shared__ float red[256];
    float s = 0.f;
    long stride = (long)gridDim.x * blockDim.x;
    for (long i = (long)blockIdx.x * blockDim.x + threadIdx.x; i < n; i += stride)
        s += fabsf(W[i]);
    red[threadIdx.x] = s;
    __syncthreads();
    for (int o = 128; o > 0; o >>= 1) {
        if ((int)threadIdx.x < o) red[threadIdx.x] += red[threadIdx.x + o];
        __syncthreads();
    }
    if (threadIdx.x == 0) partial[blockIdx.x] = red[0];
}

__global__ __launch_bounds__(256)
void finalize_scale_kernel(const float* __restrict__ partial, float numel, float* __restrict__ out) {
    __shared__ float red[256];
    float s = partial[threadIdx.x] + partial[threadIdx.x + 256] +
              partial[threadIdx.x + 512] + partial[threadIdx.x + 768];
    red[threadIdx.x] = s;
    __syncthreads();
    for (int o = 128; o > 0; o >>= 1) {
        if ((int)threadIdx.x < o) red[threadIdx.x] += red[threadIdx.x + o];
        __syncthreads();
    }
    if (threadIdx.x == 0) out[0] = fmaxf(red[0] / numel, 1e-5f);
}

__global__ __launch_bounds__(256)
void ternarize_kernel(const float* __restrict__ W, long n,
                      const float* __restrict__ wscale, signed char* __restrict__ Wt) {
    float ws = *wscale;
    long stride = (long)gridDim.x * blockDim.x;
    for (long i = (long)blockIdx.x * blockDim.x + threadIdx.x; i < n; i += stride) {
        float q = rintf(W[i] / ws);
        q = fminf(1.f, fmaxf(-1.f, q));
        Wt[i] = (signed char)q;
    }
}

__global__ __launch_bounds__(256)
void act_quant_kernel(const float* __restrict__ X, signed char* __restrict__ Xq,
                      float* __restrict__ scale, int len) {
    __shared__ float red[256];
    const long base = (long)blockIdx.x * len;
    float m = 0.f;
    for (int i = threadIdx.x; i < len; i += 256) m = fmaxf(m, fabsf(X[base + i]));
    red[threadIdx.x] = m;
    __syncthreads();
    for (int o = 128; o > 0; o >>= 1) {
        if ((int)threadIdx.x < o) red[threadIdx.x] = fmaxf(red[threadIdx.x], red[threadIdx.x + o]);
        __syncthreads();
    }
    float as = 127.f / fmaxf(red[0], 1e-5f);
    if (threadIdx.x == 0) scale[blockIdx.x] = as;
    for (int i = threadIdx.x; i < len; i += 256) {
        float q = rintf(X[base + i] * as);
        q = fminf(127.f, fmaxf(-128.f, q));
        Xq[base + i] = (signed char)q;
    }
}

__global__ __launch_bounds__(256)
void bitlinear_gemm_iu8_kernel(const signed char* __restrict__ Aq,
                               const float* __restrict__ ascale,
                               const signed char* __restrict__ W8,
                               const float* __restrict__ wscale_p,
                               int K, int mode,
                               __bf16* __restrict__ out_bf,
                               float* __restrict__ out_f) {
    __shared__ __align__(16) signed char bsh[128 * 64];

    const int tid  = threadIdx.x;
    const int wave = tid >> 5;
    const int lane = tid & 31;
    const int half = lane >> 4;
    const int l16  = lane & 15;
    const int mt   = blockIdx.x * 8 + wave;
    const int nb   = blockIdx.y;
    const float wscale = *wscale_p;

    const long arow = (long)(mt * 16 + l16) * K;

    const int nl   = tid >> 1;
    const int kseg = (tid & 1) * 32;
    const signed char* bsrc = W8 + (long)(nb * 128 + nl) * K + kseg;

    v8i acc[8];
#pragma unroll
    for (int i = 0; i < 8; i++)
#pragma unroll
        for (int j = 0; j < 8; j++) acc[i][j] = 0;

    for (int kk = 0; kk < K; kk += 64) {
        __syncthreads();
        *(int4*)(&bsh[nl * 64 + kseg])      = *(const int4*)(bsrc + kk);
        *(int4*)(&bsh[nl * 64 + kseg + 16]) = *(const int4*)(bsrc + kk + 16);
        __syncthreads();

        if (kk + 64 < K)
            __builtin_prefetch(bsrc + kk + 64, 0, 0);

        v8i af;
#pragma unroll
        for (int v = 0; v < 8; v++) {
            int ko = ((v >> 2) << 5) + (half << 4) + ((v & 3) << 2);
            af[v] = *(const int*)(Aq + arow + kk + ko);
        }

#pragma unroll
        for (int nt = 0; nt < 8; nt++) {
            const signed char* brow = &bsh[(nt * 16 + l16) * 64];
            v8i bf;
#pragma unroll
            for (int v = 0; v < 8; v++) {
                int ko = ((v >> 2) << 5) + (half << 4) + ((v & 3) << 2);
                bf[v] = *(const int*)(brow + ko);
            }
            acc[nt] = __builtin_amdgcn_wmma_i32_16x16x64_iu8(
                true, af, true, bf, acc[nt], false, false);
        }
    }

#pragma unroll
    for (int r = 0; r < 8; r++) {
        const int mrow = mt * 16 + r + half * 8;
        const float deq = wscale / ascale[mrow];
        const int b = mrow >> 11, s = mrow & 2047;
#pragma unroll
        for (int nt = 0; nt < 8; nt++) {
            const int n = nb * 128 + nt * 16 + l16;
            float f = (float)acc[nt][r] * deq;
            if (mode == 0) {
                f *= 0.08838834764831845f;
                int h = n >> 7, d = n & 127;
                out_bf[(((long)(b * NHEAD + h)) * SEQ + s) * DHEAD + d] = (__bf16)f;
            } else if (mode == 1) {
                int hk = n >> 7, d = n & 127;
                out_bf[(((long)(b * NKV + hk)) * SEQ + s) * DHEAD + d] = (__bf16)f;
            } else if (mode == 2) {
                int hk = n >> 7, d = n & 127;
                out_bf[(((long)(b * NKV + hk)) * DHEAD + d) * SEQ + s] = (__bf16)f;
            } else {
                out_f[(long)mrow * HID + n] = f;
            }
        }
    }
}

__global__ __launch_bounds__(32)
void flash_attn_kernel(const __bf16* __restrict__ Qb,
                       const __bf16* __restrict__ Kb,
                       const __bf16* __restrict__ Vtb,
                       float* __restrict__ attn) {
    __shared__ __bf16 plds[16 * 32];

    const int lane = threadIdx.x & 31;
    const int half = lane >> 4;
    const int l16  = lane & 15;
    const int qt = blockIdx.x;
    const int bh = blockIdx.y;
    const int b = bh >> 5, h = bh & 31, hk = h >> 2;

    const __bf16* Qh = Qb  + ((long)(b * NHEAD + h))  * SEQ * DHEAD;
    const __bf16* Kh = Kb  + ((long)(b * NKV  + hk)) * SEQ * DHEAD;
    const __bf16* Vh = Vtb + ((long)(b * NKV  + hk)) * DHEAD * SEQ;

    v16bf qf[4];
    {
        const __bf16* qrow = Qh + (long)(qt * 16 + l16) * DHEAD;
#pragma unroll
        for (int ks = 0; ks < 4; ks++) {
            v8u t;
#pragma unroll
            for (int v = 0; v < 8; v++) {
                int off = ks * 32 + ((v >> 2) << 4) + (half << 3) + ((v & 3) << 1);
                t[v] = *(const unsigned int*)(qrow + off);
            }
            qf[ks] = __builtin_bit_cast(v16bf, t);
        }
    }

    float mrun[8], lrun[8];
    v8f o[8];
#pragma unroll
    for (int r = 0; r < 8; r++) { mrun[r] = -1e30f; lrun[r] = 0.f; }
#pragma unroll
    for (int dt = 0; dt < 8; dt++)
#pragma unroll
        for (int r = 0; r < 8; r++) o[dt][r] = 0.f;

    const int kbmax = (qt * 16 + 15) >> 5;
    for (int kb = 0; kb <= kbmax; kb++) {
        const int kbase = kb * 32;

        v8f s0, s1;
#pragma unroll
        for (int r = 0; r < 8; r++) { s0[r] = 0.f; s1[r] = 0.f; }
#pragma unroll
        for (int ks = 0; ks < 4; ks++) {
            const __bf16* k0 = Kh + (long)(kbase + l16) * DHEAD + ks * 32;
            const __bf16* k1 = k0 + 16 * DHEAD;
            v8u t0, t1;
#pragma unroll
            for (int v = 0; v < 8; v++) {
                int off = ((v >> 2) << 4) + (half << 3) + ((v & 3) << 1);
                t0[v] = *(const unsigned int*)(k0 + off);
                t1[v] = *(const unsigned int*)(k1 + off);
            }
            s0 = __builtin_amdgcn_wmma_f32_16x16x32_bf16(
                false, qf[ks], false, __builtin_bit_cast(v16bf, t0), (short)0, s0, false, false);
            s1 = __builtin_amdgcn_wmma_f32_16x16x32_bf16(
                false, qf[ks], false, __builtin_bit_cast(v16bf, t1), (short)0, s1, false, false);
        }

#pragma unroll
        for (int r = 0; r < 8; r++) {
            const int row = qt * 16 + r + half * 8;
            if (kbase + l16 > row)      s0[r] += -1e9f;
            if (kbase + 16 + l16 > row) s1[r] += -1e9f;
            float bm = fmaxf(s0[r], s1[r]);
#pragma unroll
            for (int x = 1; x < 16; x <<= 1) bm = fmaxf(bm, __shfl_xor(bm, x, 32));
            float mnew = fmaxf(mrun[r], bm);
            float c = exp2f(mrun[r] - mnew);
            mrun[r] = mnew;
            lrun[r] *= c;
#pragma unroll
            for (int dt = 0; dt < 8; dt++) o[dt][r] *= c;
            float p0 = exp2f(s0[r] - mnew);
            float p1 = exp2f(s1[r] - mnew);
            s0[r] = p0; s1[r] = p1;
            float ps = p0 + p1;
#pragma unroll
            for (int x = 1; x < 16; x <<= 1) ps += __shfl_xor(ps, x, 32);
            lrun[r] += ps;
        }

        __syncthreads();
#pragma unroll
        for (int r = 0; r < 8; r++) {
            const int row = r + half * 8;
            plds[row * 32 + l16]      = (__bf16)s0[r];
            plds[row * 32 + 16 + l16] = (__bf16)s1[r];
        }
        __syncthreads();
        v16bf pf;
        {
            v8u t;
#pragma unroll
            for (int v = 0; v < 8; v++) {
                int off = ((v >> 2) << 4) + (half << 3) + ((v & 3) << 1);
                t[v] = *(const unsigned int*)(&plds[l16 * 32 + off]);
            }
            pf = __builtin_bit_cast(v16bf, t);
        }

#pragma unroll
        for (int dt = 0; dt < 8; dt++) {
            const __bf16* vrow = Vh + (long)(dt * 16 + l16) * SEQ + kbase;
            v8u t;
#pragma unroll
            for (int v = 0; v < 8; v++) {
                int off = ((v >> 2) << 4) + (half << 3) + ((v & 3) << 1);
                t[v] = *(const unsigned int*)(vrow + off);
            }
            o[dt] = __builtin_amdgcn_wmma_f32_16x16x32_bf16(
                false, pf, false, __builtin_bit_cast(v16bf, t), (short)0, o[dt], false, false);
        }
    }

#pragma unroll
    for (int r = 0; r < 8; r++) {
        const int row = qt * 16 + r + half * 8;
        const long tok = (long)b * SEQ + row;
        const float inv_l = 1.0f / lrun[r];
#pragma unroll
        for (int dt = 0; dt < 8; dt++)
            attn[tok * HID + h * DHEAD + dt * 16 + l16] = o[dt][r] * inv_l;
    }
}

extern "C" void kernel_launch(void* const* d_in, const int* in_sizes, int n_in,
                              void* d_out, int out_size, void* d_ws, size_t ws_size,
                              hipStream_t stream) {
    const float* x    = (const float*)d_in[0];
    const float* Wq   = (const float*)d_in[2];
    const float* Wk   = (const float*)d_in[3];
    const float* Wv   = (const float*)d_in[4];
    const float* Wo   = (const float*)d_in[5];

    size_t off = 0;
    auto take = [&](size_t bytes) -> void* {
        off = (off + 255) & ~(size_t)255;
        void* p = (char*)d_ws + off;
        off += bytes;
        return p;
    };

    float* partial  = (float*)take(4 * 1024 * sizeof(float));
    float* wscales  = (float*)take(4 * sizeof(float));
    float* as_x     = (float*)take(TOK * sizeof(float));
    signed char* xq = (signed char*)take((size_t)TOK * HID);
    signed char* wq8 = (signed char*)take((size_t)HID * HID);
    signed char* wk8 = (signed char*)take((size_t)KVW * HID);
    signed char* wv8 = (signed char*)take((size_t)KVW * HID);
    signed char* wo8 = (signed char*)take((size_t)HID * HID);
    __bf16* Qb  = (__bf16*)take((size_t)TOK * HID * 2);
    __bf16* Kb  = (__bf16*)take((size_t)TOK * KVW * 2);
    __bf16* Vtb = (__bf16*)take((size_t)TOK * KVW * 2);
    float*  attn = (float*)take((size_t)TOK * HID * sizeof(float));

    const long nQ = (long)HID * HID, nK = (long)KVW * HID;

    abs_partial_kernel<<<1024, 256, 0, stream>>>(Wq, nQ, partial + 0 * 1024);
    finalize_scale_kernel<<<1, 256, 0, stream>>>(partial + 0 * 1024, (float)nQ, wscales + 0);
    abs_partial_kernel<<<1024, 256, 0, stream>>>(Wk, nK, partial + 1 * 1024);
    finalize_scale_kernel<<<1, 256, 0, stream>>>(partial + 1 * 1024, (float)nK, wscales + 1);
    abs_partial_kernel<<<1024, 256, 0, stream>>>(Wv, nK, partial + 2 * 1024);
    finalize_scale_kernel<<<1, 256, 0, stream>>>(partial + 2 * 1024, (float)nK, wscales + 2);
    abs_partial_kernel<<<1024, 256, 0, stream>>>(Wo, nQ, partial + 3 * 1024);
    finalize_scale_kernel<<<1, 256, 0, stream>>>(partial + 3 * 1024, (float)nQ, wscales + 3);

    ternarize_kernel<<<2048, 256, 0, stream>>>(Wq, nQ, wscales + 0, wq8);
    ternarize_kernel<<<1024, 256, 0, stream>>>(Wk, nK, wscales + 1, wk8);
    ternarize_kernel<<<1024, 256, 0, stream>>>(Wv, nK, wscales + 2, wv8);
    ternarize_kernel<<<2048, 256, 0, stream>>>(Wo, nQ, wscales + 3, wo8);

    act_quant_kernel<<<TOK, 256, 0, stream>>>(x, xq, as_x, HID);

    bitlinear_gemm_iu8_kernel<<<dim3(TOK / 128, HID / 128), 256, 0, stream>>>(
        xq, as_x, wq8, wscales + 0, HID, 0, Qb, nullptr);
    bitlinear_gemm_iu8_kernel<<<dim3(TOK / 128, KVW / 128), 256, 0, stream>>>(
        xq, as_x, wk8, wscales + 1, HID, 1, Kb, nullptr);
    bitlinear_gemm_iu8_kernel<<<dim3(TOK / 128, KVW / 128), 256, 0, stream>>>(
        xq, as_x, wv8, wscales + 2, HID, 2, Vtb, nullptr);

    flash_attn_kernel<<<dim3(SEQ / 16, 2 * NHEAD), 32, 0, stream>>>(Qb, Kb, Vtb, attn);

    act_quant_kernel<<<TOK, 256, 0, stream>>>(attn, xq, as_x, HID);
    bitlinear_gemm_iu8_kernel<<<dim3(TOK / 128, HID / 128), 256, 0, stream>>>(
        xq, as_x, wo8, wscales + 3, HID, 3, nullptr, (float*)d_out);

    (void)in_sizes; (void)n_in; (void)out_size; (void)ws_size;
}
